// GroupedQueryAttention_74259984547851
// MI455X (gfx1250) — matrix operand order repaired and checked by reading, not compiled, not run on hardware
//
#include <hip/hip_runtime.h>
#include <hip/hip_bf16.h>

typedef __attribute__((ext_vector_type(16))) __bf16 v16bf;
typedef __attribute__((ext_vector_type(8)))  float  v8f;

union FragB16 {
    v16bf v;
    unsigned int u[8];
};

__device__ __forceinline__ unsigned short f2bf(float f) {
    union { float f; unsigned int u; } x;
    x.f = f;
    unsigned int u = x.u;
    u += 0x7FFFu + ((u >> 16) & 1u);   // round-to-nearest-even
    return (unsigned short)(u >> 16);
}

__device__ __forceinline__ v8f zero8() {
    v8f z = {0.f, 0.f, 0.f, 0.f, 0.f, 0.f, 0.f, 0.f};
    return z;
}

// ---------------------------------------------------------------------------
// fp32 -> bf16 conversion
// ---------------------------------------------------------------------------
__global__ void __launch_bounds__(256, 1)
convert_f32_bf16(const float* __restrict__ in,
                 unsigned short* __restrict__ out, int n) {
    int i = blockIdx.x * blockDim.x + threadIdx.x;
    if (i < n) out[i] = f2bf(in[i]);
}

// ---------------------------------------------------------------------------
// A fragment: 16x32 bf16 (M x K), X row-major, contiguous k order (this code's, not the ISA table order)
//   lane = M, VGPR j holds K pair k0 = (lane/16)*16 + 2j
// ---------------------------------------------------------------------------
__device__ __forceinline__ void load_frag_A(const unsigned short* __restrict__ X,
                                            int row_base, int ldx, int k_base,
                                            FragB16& f) {
    const int lane = threadIdx.x & 31;
    const int m    = lane & 15;
    const int half = lane >> 4;
    const unsigned short* p = X + (size_t)(row_base + m) * ldx + k_base;
#pragma unroll
    for (int j = 0; j < 8; ++j) {
        int k0 = (half << 4) + (j << 1);
        f.u[j] = *(const unsigned int*)(p + k0);
    }
}

// ---------------------------------------------------------------------------
// B fragment: 32x16 bf16 (K x N). Element (k,n) = W[(col_base+n)*ldw + k_base + k]
//   lane = N; lanes 0..15 hold K=0..15, lanes 16..31 hold K=16..31;
//   VGPR j holds K pair k0 = (lane/16)*16 + 2j
// ---------------------------------------------------------------------------
__device__ __forceinline__ void load_frag_B(const unsigned short* __restrict__ W,
                                            int col_base, int ldw, int k_base,
                                            FragB16& f) {
    const int lane = threadIdx.x & 31;
    const int n    = lane & 15;
    const int half = lane >> 4;
    const unsigned short* p = W + (size_t)(col_base + n) * ldw + k_base + (half << 4);
#pragma unroll
    for (int j = 0; j < 8; ++j) {
        f.u[j] = *(const unsigned int*)(p + (j << 1));
    }
}

// ---------------------------------------------------------------------------
// Generic WMMA GEMM:  y[m,n] = (sum_k X[m,k]*W[n,k] + bias[n]) * outScale
//   X: bf16 [M,K] row-major,  W: bf16 [N,K] row-major (B = W^T)
//   OUTMODE 0: bf16 out[m*N+n]   1: bf16 out[n*M+m] (transposed)   2: f32 out[m*N+n]
// blockDim = 128 (4 waves); wave w covers rows (blockIdx.y*4+w)*16..+15,
// cols blockIdx.x*64..+63.  A-fragment double-buffered so next k-block's loads
// overlap the current block's 4 WMMAs.
// ---------------------------------------------------------------------------
template <int OUTMODE>
__global__ void __launch_bounds__(128, 1)
gemm_bf16_wmma(const unsigned short* __restrict__ X,
               const unsigned short* __restrict__ W,
               const float* __restrict__ bias,
               void* __restrict__ out,
               int M, int N, int K, float outScale) {
    const int wave  = threadIdx.x >> 5;
    const int lane  = threadIdx.x & 31;
    const int mBase = (blockIdx.y * 4 + wave) * 16;
    const int nBase = blockIdx.x * 64;

    v8f acc[4];
#pragma unroll
    for (int t = 0; t < 4; ++t) acc[t] = zero8();

    FragB16 a;
    load_frag_A(X, mBase, K, 0, a);

    for (int k0 = 0; k0 < K; k0 += 32) {
        FragB16 b[4];
#pragma unroll
        for (int t = 0; t < 4; ++t)
            load_frag_B(W, nBase + t * 16, K, k0, b[t]);

        FragB16 a_next = a;
        if (k0 + 32 < K) load_frag_A(X, mBase, K, k0 + 32, a_next);

#pragma unroll
        for (int t = 0; t < 4; ++t) {
            acc[t] = __builtin_amdgcn_wmma_f32_16x16x32_bf16(
                false, a.v, false, b[t].v, (short)0, acc[t], false, false);
        }
        a = a_next;
    }

    const int half = lane >> 4;
    const int nloc = lane & 15;
#pragma unroll
    for (int t = 0; t < 4; ++t) {
        const int col = nBase + t * 16 + nloc;
        const float bv = bias[col];
#pragma unroll
        for (int j = 0; j < 8; ++j) {
            const int row = mBase + j + 8 * half;
            const float v = (acc[t][j] + bv) * outScale;
            if (OUTMODE == 2) {
                ((float*)out)[(size_t)row * N + col] = v;
            } else if (OUTMODE == 0) {
                ((unsigned short*)out)[(size_t)row * N + col] = f2bf(v);
            } else {
                ((unsigned short*)out)[(size_t)col * M + row] = f2bf(v);
            }
        }
    }
}

// ---------------------------------------------------------------------------
// Flash-style GQA attention.
//   Qb: bf16 [S, 2048], pre-scaled by 1/sqrt(HD); head hh -> cols hh*64..+63
//   Kb: bf16 [S, 512];  group g -> cols g*64..+63
//   Vt: bf16 [512, S]   (V transposed: Vt[(g*64+d)*S + s])
//   Ob: bf16 [S, 2048]  pre-permuted to channel c = (hh%4)*512 + (hh/4)*64 + d
// One wave per (16-query block, head). Per 32-key block:
//   - 4 K-fragment loads issued together, then 4 score WMMAs
//   - 4 V-fragment loads issued BEFORE softmax (latency hidden under expf work)
//   - row max via ds_bpermute tree; row SUM via one extra WMMA against an
//     all-ones B fragment (replaces 32 ds_bpermute per block)
//   - P transposed C-layout -> A-layout through 1KB LDS, 4 PV WMMAs
// ---------------------------------------------------------------------------
__global__ void __launch_bounds__(32, 1)
gqa_flash_wmma(const unsigned short* __restrict__ Qb,
               const unsigned short* __restrict__ Kb,
               const unsigned short* __restrict__ Vt,
               unsigned short* __restrict__ Ob) {
    __shared__ unsigned short Plds[16 * 32];

    const int lane = threadIdx.x & 31;
    const int qb   = blockIdx.x;           // 0..127 : 16-query block
    const int hh   = blockIdx.y;           // 0..31  : query head
    const int g    = hh >> 2;              // KV group (HPG = 4)
    const int obase = (hh & 3) * 512 + g * 64;   // '(h g d)' channel layout
    const int m    = lane & 15;
    const int half = lane >> 4;

    // all-ones bf16 B fragment (for row-sum WMMA): 1.0bf16 = 0x3F80
    FragB16 ones;
#pragma unroll
    for (int j = 0; j < 8; ++j) ones.u[j] = 0x3F803F80u;

    // Preload Q A-fragments (head-dim 64 = two K=32 chunks)
    FragB16 qa[2];
    {
        const unsigned short* qp = Qb + (size_t)(qb * 16 + m) * 2048 + hh * 64;
#pragma unroll
        for (int c = 0; c < 2; ++c) {
#pragma unroll
            for (int j = 0; j < 8; ++j) {
                int k0 = c * 32 + (half << 4) + (j << 1);
                qa[c].u[j] = *(const unsigned int*)(qp + k0);
            }
        }
    }

    v8f acc[4];
#pragma unroll
    for (int t = 0; t < 4; ++t) acc[t] = zero8();
    float m_i[8], l_i[8];
#pragma unroll
    for (int j = 0; j < 8; ++j) { m_i[j] = -1e30f; l_i[j] = 0.f; }

    for (int kb = 0; kb < 64; ++kb) {       // 64 key blocks of 32 keys
        const int keyBase = kb * 32;

        // prefetch next key block's K and V rows into the cache hierarchy
        if (kb + 1 < 64) {
            const unsigned short* kpre =
                Kb + (size_t)(keyBase + 32 + m) * 512 + g * 64 + (half << 4);
            __builtin_prefetch(kpre, 0, 3);
            const unsigned short* vpre =
                Vt + (size_t)(g * 64 + m) * 2048 + keyBase + 32 + (half << 4);
            __builtin_prefetch(vpre, 0, 3);
        }

        // ---- issue all 4 K B-fragment loads, then 4 score WMMAs ----
        FragB16 kf[2][2];   // [tile][k-chunk]
#pragma unroll
        for (int t = 0; t < 2; ++t) {
            const unsigned short* kp =
                Kb + (size_t)(keyBase + t * 16 + m) * 512 + g * 64 + (half << 4);
#pragma unroll
            for (int j = 0; j < 8; ++j) {
                kf[t][0].u[j] = *(const unsigned int*)(kp + (j << 1));
                kf[t][1].u[j] = *(const unsigned int*)(kp + 32 + (j << 1));
            }
        }
        v8f s[2];
#pragma unroll
        for (int t = 0; t < 2; ++t) {
            s[t] = __builtin_amdgcn_wmma_f32_16x16x32_bf16(
                false, qa[0].v, false, kf[t][0].v, (short)0, zero8(), false, false);
            s[t] = __builtin_amdgcn_wmma_f32_16x16x32_bf16(
                false, qa[1].v, false, kf[t][1].v, (short)0, s[t], false, false);
        }

        // ---- issue V B-fragment loads early; latency hides under softmax ----
        FragB16 vb[4];
#pragma unroll
        for (int t4 = 0; t4 < 4; ++t4) {
            const unsigned short* vp =
                Vt + (size_t)(g * 64 + t4 * 16 + m) * 2048 + keyBase + (half << 4);
#pragma unroll
            for (int j = 0; j < 8; ++j)
                vb[t4].u[j] = *(const unsigned int*)(vp + (j << 1));
        }

        // ---- online softmax (row = j + 8*half, owned by one 16-lane half) ----
        float alpha_a[8];
#pragma unroll
        for (int j = 0; j < 8; ++j) {
            const float s0 = s[0][j];
            const float s1 = s[1][j];
            float mx = fmaxf(s0, s1);
            mx = fmaxf(mx, __shfl_xor(mx, 1, 32));
            mx = fmaxf(mx, __shfl_xor(mx, 2, 32));
            mx = fmaxf(mx, __shfl_xor(mx, 4, 32));
            mx = fmaxf(mx, __shfl_xor(mx, 8, 32));
            const float mn    = fmaxf(m_i[j], mx);
            const float alpha = __expf(m_i[j] - mn);
            alpha_a[j] = alpha;
            m_i[j] = mn;
            const float p0 = __expf(s0 - mn);
            const float p1 = __expf(s1 - mn);
#pragma unroll
            for (int t4 = 0; t4 < 4; ++t4) acc[t4][j] *= alpha;
            const int row = j + 8 * half;
            Plds[row * 32 + m]      = f2bf(p0);
            Plds[row * 32 + 16 + m] = f2bf(p1);
        }
        __syncthreads();

        // ---- read P back as A fragment (16x32, K = keys-in-block) ----
        FragB16 pa;
        {
            const unsigned int* pl = (const unsigned int*)Plds;
#pragma unroll
            for (int j = 0; j < 8; ++j) {
                int k0 = (half << 4) + (j << 1);
                pa.u[j] = pl[(m * 32 + k0) >> 1];
            }
        }
        __syncthreads();

        // ---- row sums in one WMMA: D(m,n) = sum_k P(m,k) * 1 ----
        v8f rsum = __builtin_amdgcn_wmma_f32_16x16x32_bf16(
            false, pa.v, false, ones.v, (short)0, zero8(), false, false);
#pragma unroll
        for (int j = 0; j < 8; ++j)
            l_i[j] = l_i[j] * alpha_a[j] + rsum[j];

        // ---- out += P @ V  (4 N-tiles of 16 over head-dim 64) ----
#pragma unroll
        for (int t4 = 0; t4 < 4; ++t4) {
            acc[t4] = __builtin_amdgcn_wmma_f32_16x16x32_bf16(
                false, pa.v, false, vb[t4].v, (short)0, acc[t4], false, false);
        }
    }

    // ---- normalize and store (already in '(h g d)' channel order) ----
#pragma unroll
    for (int t4 = 0; t4 < 4; ++t4) {
#pragma unroll
        for (int j = 0; j < 8; ++j) {
            const int row = qb * 16 + j + 8 * half;
            const float v = acc[t4][j] / l_i[j];
            Ob[(size_t)row * 2048 + obase + t4 * 16 + m] = f2bf(v);
        }
    }
}

// ---------------------------------------------------------------------------
// launch
// ---------------------------------------------------------------------------
extern "C" void kernel_launch(void* const* d_in, const int* in_sizes, int n_in,
                              void* d_out, int out_size, void* d_ws, size_t ws_size,
                              hipStream_t stream) {
    const float* x  = (const float*)d_in[0];
    const float* Wq = (const float*)d_in[1];
    const float* bq = (const float*)d_in[2];
    const float* Wk = (const float*)d_in[3];
    const float* bk = (const float*)d_in[4];
    const float* Wv = (const float*)d_in[5];
    const float* bv = (const float*)d_in[6];
    const float* Wo = (const float*)d_in[7];
    const float* bo = (const float*)d_in[8];

    char* ws = (char*)d_ws;
    const size_t MB = 1024ull * 1024ull;
    unsigned short* Xb  = (unsigned short*)(ws + 0 * MB);   // [2048,2048] bf16
    unsigned short* Qb  = (unsigned short*)(ws + 8 * MB);   // [2048,2048] bf16
    unsigned short* Ab  = (unsigned short*)(ws + 16 * MB);  // attn out [2048,2048] bf16
    unsigned short* Wqb = (unsigned short*)(ws + 24 * MB);  // [2048,2048] bf16
    unsigned short* Wob = (unsigned short*)(ws + 32 * MB);  // [2048,2048] bf16
    unsigned short* Wkb = (unsigned short*)(ws + 40 * MB);  // [512,2048] bf16
    unsigned short* Wvb = (unsigned short*)(ws + 42 * MB);  // [512,2048] bf16
    unsigned short* Kb  = (unsigned short*)(ws + 44 * MB);  // [2048,512] bf16
    unsigned short* Vt  = (unsigned short*)(ws + 46 * MB);  // [512,2048] bf16 (V^T)

    const int nFull = 2048 * 2048;
    const int nKV   = 512 * 2048;
    convert_f32_bf16<<<(nFull + 255) / 256, 256, 0, stream>>>(x,  Xb,  nFull);
    convert_f32_bf16<<<(nFull + 255) / 256, 256, 0, stream>>>(Wq, Wqb, nFull);
    convert_f32_bf16<<<(nKV  + 255) / 256, 256, 0, stream>>>(Wk, Wkb, nKV);
    convert_f32_bf16<<<(nKV  + 255) / 256, 256, 0, stream>>>(Wv, Wvb, nKV);
    convert_f32_bf16<<<(nFull + 255) / 256, 256, 0, stream>>>(Wo, Wob, nFull);

    // Q = (Xb @ Wq^T + bq) * (1/sqrt(64))  -> bf16 [2048,2048]
    gemm_bf16_wmma<0><<<dim3(32, 32), 128, 0, stream>>>(Xb, Wqb, bq, Qb,
                                                        2048, 2048, 2048, 0.125f);
    // K = Xb @ Wk^T + bk  -> bf16 [2048,512]
    gemm_bf16_wmma<0><<<dim3(8, 32), 128, 0, stream>>>(Xb, Wkb, bk, Kb,
                                                       2048, 512, 2048, 1.0f);
    // V = Xb @ Wv^T + bv  -> bf16 transposed [512,2048]
    gemm_bf16_wmma<1><<<dim3(8, 32), 128, 0, stream>>>(Xb, Wvb, bv, Vt,
                                                       2048, 512, 2048, 1.0f);

    // flash GQA: 128 query blocks x 32 heads, one wave each
    gqa_flash_wmma<<<dim3(128, 32), 32, 0, stream>>>(Qb, Kb, Vt, Ab);

    // out = Ab @ Wo^T + bo -> fp32 [2048,2048]
    gemm_bf16_wmma<2><<<dim3(32, 32), 128, 0, stream>>>(Ab, Wob, bo, (float*)d_out,
                                                        2048, 2048, 2048, 1.0f);
}
